// CFConv_44332652429581
// MI455X (gfx1250) — matrix operand order repaired and checked by reading, not compiled, not run on hardware
//
#include <hip/hip_runtime.h>
#include <math.h>

// ---------------------------------------------------------------------------
// SchNet CFConv, fused for MI455X (gfx1250, wave32, WMMA f32_16x16x32_f16).
//   K0: one-time prep: transposed, f16, LDS-stride-padded weight images in ws
//   K1: y = x @ W_in2f                         (WMMA GEMM, 2048x128x128)
//   K2: per-(b,a) fused filter net + cutoff + gather + neighbor reduction
//       H  = ssp(f_ij @ W1 + b1)               (WMMA, K=32 padded from 25)
//       Wf = H @ W2 + b2                       (WMMA, K=128)
//       agg[f] = sum_n Wf[n,f]*C[n]*mask[n]*y[b,nbr[n],f]   (epilogue-fused)
//   K3: out = ssp(agg @ W_out + b_out)         (WMMA GEMM, 2048x128x128)
// Weight tiles are staged into LDS with GLOBAL_LOAD_ASYNC_TO_LDS_B128
// (ASYNCcnt) so staging overlaps the f32->f16 conversion of activations.
// Workspace: y (1 MB f32) + agg (1 MB f32) + f16 weight images (~112 KB).
// ---------------------------------------------------------------------------

typedef __attribute__((ext_vector_type(16))) _Float16 v16h;
typedef __attribute__((ext_vector_type(8)))  _Float16 v8h;
typedef __attribute__((ext_vector_type(8)))  float    v8f;

#define NB 8
#define NA 256
#define NBH_ 128
#define NF 128
#define NG 25
#define LDF 40    // padded K-stride (halfs) for K=32 tiles, keeps 16B align
#define LDH 136   // padded K-stride (halfs) for K=128 tiles

// workspace layout (element offsets)
#define YWS_F32    (NB * NA * NF)          // 262144 floats
#define AGG_F32    (NB * NA * NF)          // 262144 floats
#define WINT_OFF   0                        // halfs: 128*LDH
#define WOUTT_OFF  (128 * LDH)              // halfs: 128*LDH
#define W2T_OFF    (2 * 128 * LDH)          // halfs: 128*LDH
#define W1T_OFF    (3 * 128 * LDH)          // halfs: 128*LDF

__device__ __forceinline__ float sspf(float v) {
  float sp = fmaxf(v, 0.f) + log1pf(expf(-fabsf(v)));
  return sp - 0.69314718055994531f;
}

__device__ __forceinline__ v16h frag_cat(v8h lo, v8h hi) {
  return __builtin_shufflevector(lo, hi, 0, 1, 2, 3, 4, 5, 6, 7,
                                 8, 9, 10, 11, 12, 13, 14, 15);
}

// A fragment (16x32 f16, MxK): row-major LDS tile, stride ld halfs.
__device__ __forceinline__ v16h load_frag_a(const _Float16* s, int row0,
                                            int ld, int k0) {
  int lane = threadIdx.x & 31;
  int m = lane & 15;
  int kb = k0 + ((lane >> 4) << 3);
  const _Float16* p = s + (row0 + m) * ld + kb;
  v8h lo = *reinterpret_cast<const v8h*>(p);
  v8h hi = *reinterpret_cast<const v8h*>(p + 16);
  return frag_cat(lo, hi);
}

// B fragment (32x16 f16, KxN): LDS holds B transposed [N x ld].
__device__ __forceinline__ v16h load_frag_b(const _Float16* s, int n0,
                                            int ld, int k0) {
  int lane = threadIdx.x & 31;
  int n = lane & 15;
  int kb = k0 + ((lane >> 4) << 3);
  const _Float16* p = s + (n0 + n) * ld + kb;
  v8h lo = *reinterpret_cast<const v8h*>(p);
  v8h hi = *reinterpret_cast<const v8h*>(p + 16);
  return frag_cat(lo, hi);
}

__device__ __forceinline__ v8f wmma16(v16h a, v16h b, v8f c) {
  return __builtin_amdgcn_wmma_f32_16x16x32_f16(false, a, false, b,
                                                (short)0, c, false, false);
}

// CDNA5 async global->LDS copy (16B per lane), tracked by ASYNCcnt.
__device__ __forceinline__ void async_copy16(unsigned lds_off,
                                             const void* gaddr) {
  unsigned long long ga = (unsigned long long)gaddr;
  asm volatile("global_load_async_to_lds_b128 %0, %1, off"
               :: "v"(lds_off), "v"(ga) : "memory");
}
__device__ __forceinline__ void wait_async0() {
  asm volatile("s_wait_asynccnt 0x0" ::: "memory");
}
// LDS byte offset of a shared-memory pointer (flat LDS aperture keeps the
// LDS offset in the low 32 bits).
__device__ __forceinline__ unsigned lds_off_of(const void* p) {
  return (unsigned)(unsigned long long)p;
}

// ---------------------------------------------------------------------------
// K0: one-time weight prep: transposed f16 images, padded to LDS strides.
// grid = 128 blocks x 256 threads; block r handles row r of each image.
// ---------------------------------------------------------------------------
__global__ __launch_bounds__(256) void k_prep(const float* __restrict__ w_in2f,
                                              const float* __restrict__ w1,
                                              const float* __restrict__ w2,
                                              const float* __restrict__ w_out,
                                              _Float16* __restrict__ wimg) {
  int r = blockIdx.x;        // output row (the N index of the original matrix)
  int tid = threadIdx.x;
  if (tid < LDH) {
    float vi = (tid < 128) ? w_in2f[tid * NF + r] : 0.f;
    float vo = (tid < 128) ? w_out[tid * NF + r] : 0.f;
    float v2 = (tid < 128) ? w2[tid * NF + r] : 0.f;
    wimg[WINT_OFF + r * LDH + tid] = (_Float16)vi;
    wimg[WOUTT_OFF + r * LDH + tid] = (_Float16)vo;
    wimg[W2T_OFF + r * LDH + tid] = (_Float16)v2;
  }
  if (tid < LDF) {
    float v1 = (tid < NG) ? w1[tid * NF + r] : 0.f;
    wimg[W1T_OFF + r * LDF + tid] = (_Float16)v1;
  }
}

// ---------------------------------------------------------------------------
// K1: y[2048,128] = x[2048,128] @ W_in2f[128,128]   (no bias, f32 out)
// ---------------------------------------------------------------------------
__global__ __launch_bounds__(256) void k_gemm_y(const float* __restrict__ x,
                                                const _Float16* __restrict__ wt,
                                                float* __restrict__ y) {
  __shared__ alignas(16) _Float16 Xs[128 * LDH];
  __shared__ alignas(16) _Float16 Ws[128 * LDH];
  int tid = threadIdx.x;
  int row0 = blockIdx.x * 128;

  // async-copy W^T (f16, pre-padded) straight into LDS: 128*LDH*2 B
  {
    unsigned base = lds_off_of(Ws);
    for (int i = tid; i < (128 * LDH * 2) / 16; i += 256)
      async_copy16(base + i * 16, (const char*)wt + i * 16);
  }
  // stage X with f32->f16 conversion (overlaps async weight copy)
  for (int i = tid; i < 128 * 128; i += 256) {
    int r = i >> 7, c = i & 127;
    Xs[r * LDH + c] = (_Float16)x[(row0 + r) * 128 + c];
  }
  wait_async0();
  __syncthreads();

  int wv = tid >> 5, lane = tid & 31;
  int mrow = wv * 16;
  v8f acc[8] = {};
#pragma unroll
  for (int k0 = 0; k0 < 128; k0 += 32) {
    v16h af = load_frag_a(Xs, mrow, LDH, k0);
#pragma unroll
    for (int c = 0; c < 8; ++c) {
      v16h bf = load_frag_b(Ws, c * 16, LDH, k0);
      acc[c] = wmma16(af, bf, acc[c]);
    }
  }
  int nlo = lane & 15, mhi = (lane >> 4) << 3;
#pragma unroll
  for (int c = 0; c < 8; ++c) {
    union { v8f v; float f[8]; } u; u.v = acc[c];
#pragma unroll
    for (int r = 0; r < 8; ++r) {
      int m = row0 + mrow + r + mhi;
      y[m * 128 + c * 16 + nlo] = u.f[r];
    }
  }
}

// ---------------------------------------------------------------------------
// K2: fused filter network + cutoff + gather + neighbor-sum per (b,a)
// ---------------------------------------------------------------------------
__global__ __launch_bounds__(256) void k_cfconv(
    const float* __restrict__ fij,   // (B,A,NBH,25)
    const float* __restrict__ rij,   // (B,A,NBH)
    const int*   __restrict__ nbrs,  // (B,A,NBH)
    const float* __restrict__ mask,  // (B,A,NBH)
    const _Float16* __restrict__ w1t,  // [128*LDF] f16 image
    const float* __restrict__ b1,    // (128)
    const _Float16* __restrict__ w2t,  // [128*LDH] f16 image
    const float* __restrict__ b2,    // (128)
    const float* __restrict__ yws,   // (B*A,128)
    float* __restrict__ aggws) {     // (B*A,128)
  __shared__ alignas(16) _Float16 Fs[NBH_ * LDF];   // f_ij tile, K padded
  __shared__ alignas(16) _Float16 W1s[NF * LDF];    // W1^T, K padded
  __shared__ alignas(16) _Float16 Hs[NBH_ * LDH];   // hidden, f16
  __shared__ alignas(16) _Float16 W2s[NF * LDH];    // W2^T
  __shared__ float Cc[NBH_];
  __shared__ float b1s[NF], b2s[NF];
  __shared__ int   nbrL[NBH_];
  __shared__ float aggL[NF];

  int tid = threadIdx.x;
  int g = blockIdx.x;        // b*A + a
  int bidx = g >> 8;         // A = 256
  int pairBase = g * NBH_;

  // async-copy weight images into LDS (overlaps f_ij conversion below)
  {
    unsigned base1 = lds_off_of(W1s);
    for (int i = tid; i < (NF * LDF * 2) / 16; i += 256)
      async_copy16(base1 + i * 16, (const char*)w1t + i * 16);
    unsigned base2 = lds_off_of(W2s);
    for (int i = tid; i < (NF * LDH * 2) / 16; i += 256)
      async_copy16(base2 + i * 16, (const char*)w2t + i * 16);
  }

  // stage f_ij (zero-pad K 25->32+) with f32->f16 conversion
  for (int i = tid; i < NBH_ * LDF; i += 256) {
    int n = i / LDF, k = i - n * LDF;
    float v = (k < NG) ? fij[(pairBase + n) * NG + k] : 0.f;
    Fs[i] = (_Float16)v;
  }
  if (tid < NBH_) {
    float r = rij[pairBase + tid];
    float c = 0.5f * (__cosf(r * 0.62831853071795864f) + 1.f) *
              (r < 5.f ? 1.f : 0.f);
    Cc[tid] = c * mask[pairBase + tid];
    b1s[tid] = b1[tid];
    b2s[tid] = b2[tid];
    nbrL[tid] = nbrs[pairBase + tid];
    aggL[tid] = 0.f;
  }
  wait_async0();
  __syncthreads();

  int wv = tid >> 5, lane = tid & 31;
  int mrow = wv * 16;                 // this wave's 16 neighbor rows
  int nlo = lane & 15, mhi = (lane >> 4) << 3;

  // GEMM1: H = ssp(F @ W1 + b1), single K=32 step
  {
    v16h af = load_frag_a(Fs, mrow, LDF, 0);
#pragma unroll
    for (int c = 0; c < 8; ++c) {
      v16h bf = load_frag_b(W1s, c * 16, LDF, 0);
      v8f acc = {};
      acc = wmma16(af, bf, acc);
      union { v8f v; float f[8]; } u; u.v = acc;
      int n = c * 16 + nlo;
#pragma unroll
      for (int r = 0; r < 8; ++r) {
        int m = mrow + r + mhi;
        Hs[m * LDH + n] = (_Float16)sspf(u.f[r] + b1s[n]);
      }
    }
  }
  __syncthreads();

  // GEMM2: Wf = H @ W2 + b2; epilogue fuses cutoff*mask, neighbor gather of
  // y, and the reduction over the neighbor axis into LDS atomics.
  int ybase = bidx * NA * NF;
  v8f acc[8] = {};
#pragma unroll
  for (int k0 = 0; k0 < 128; k0 += 32) {
    v16h af = load_frag_a(Hs, mrow, LDH, k0);
#pragma unroll
    for (int c = 0; c < 8; ++c) {
      v16h bf = load_frag_b(W2s, c * 16, LDH, k0);
      acc[c] = wmma16(af, bf, acc[c]);
    }
  }
#pragma unroll
  for (int c = 0; c < 8; ++c) {
    union { v8f v; float f[8]; } u; u.v = acc[c];
    int n = c * 16 + nlo;
    float part = 0.f;
#pragma unroll
    for (int r = 0; r < 8; ++r) {
      int m = mrow + r + mhi;                       // neighbor row
      float wfil = (u.f[r] + b2s[n]) * Cc[m];
      float yv = yws[ybase + nbrL[m] * NF + n];     // gathered embedding
      part += wfil * yv;
    }
    atomicAdd(&aggL[n], part);                      // ds_add_f32
  }
  __syncthreads();
  if (tid < NF) aggws[g * NF + tid] = aggL[tid];
}

// ---------------------------------------------------------------------------
// K3: out = ssp(agg[2048,128] @ W_out[128,128] + b_out)
// ---------------------------------------------------------------------------
__global__ __launch_bounds__(256) void k_gemm_out(const float* __restrict__ a,
                                                  const _Float16* __restrict__ wt,
                                                  const float* __restrict__ bo,
                                                  float* __restrict__ out) {
  __shared__ alignas(16) _Float16 As[128 * LDH];
  __shared__ alignas(16) _Float16 Ws[128 * LDH];
  __shared__ float bos[128];
  int tid = threadIdx.x;
  int row0 = blockIdx.x * 128;

  {
    unsigned base = lds_off_of(Ws);
    for (int i = tid; i < (128 * LDH * 2) / 16; i += 256)
      async_copy16(base + i * 16, (const char*)wt + i * 16);
  }
  for (int i = tid; i < 128 * 128; i += 256) {
    int r = i >> 7, c = i & 127;
    As[r * LDH + c] = (_Float16)a[(row0 + r) * 128 + c];
  }
  if (tid < 128) bos[tid] = bo[tid];
  wait_async0();
  __syncthreads();

  int wv = tid >> 5, lane = tid & 31;
  int mrow = wv * 16;
  v8f acc[8] = {};
#pragma unroll
  for (int k0 = 0; k0 < 128; k0 += 32) {
    v16h af = load_frag_a(As, mrow, LDH, k0);
#pragma unroll
    for (int c = 0; c < 8; ++c) {
      v16h bf = load_frag_b(Ws, c * 16, LDH, k0);
      acc[c] = wmma16(af, bf, acc[c]);
    }
  }
  int nlo = lane & 15, mhi = (lane >> 4) << 3;
#pragma unroll
  for (int c = 0; c < 8; ++c) {
    union { v8f v; float f[8]; } u; u.v = acc[c];
    int n = c * 16 + nlo;
#pragma unroll
    for (int r = 0; r < 8; ++r) {
      int m = row0 + mrow + r + mhi;
      out[m * 128 + n] = sspf(u.f[r] + bos[n]);
    }
  }
}

// ---------------------------------------------------------------------------
extern "C" void kernel_launch(void* const* d_in, const int* in_sizes, int n_in,
                              void* d_out, int out_size, void* d_ws,
                              size_t ws_size, hipStream_t stream) {
  const float* x      = (const float*)d_in[0];
  const float* rij    = (const float*)d_in[1];
  const int*   nbrs   = (const int*)d_in[2];
  const float* mask   = (const float*)d_in[3];
  const float* fij    = (const float*)d_in[4];
  const float* w_in2f = (const float*)d_in[5];
  const float* w1     = (const float*)d_in[6];
  const float* b1     = (const float*)d_in[7];
  const float* w2     = (const float*)d_in[8];
  const float* b2     = (const float*)d_in[9];
  const float* w_out  = (const float*)d_in[10];
  const float* b_out  = (const float*)d_in[11];
  float* out = (float*)d_out;

  float* yws   = (float*)d_ws;            // 1 MB
  float* aggws = yws + YWS_F32;           // 1 MB
  _Float16* wimg = (_Float16*)(aggws + AGG_F32);  // ~112 KB f16 weight images

  k_prep<<<128, 256, 0, stream>>>(w_in2f, w1, w2, w_out, wimg);
  k_gemm_y<<<16, 256, 0, stream>>>(x, wimg + WINT_OFF, yws);
  k_cfconv<<<NB * NA, 256, 0, stream>>>(fij, rij, nbrs, mask,
                                        wimg + W1T_OFF, b1,
                                        wimg + W2T_OFF, b2, yws, aggws);
  k_gemm_out<<<16, 256, 0, stream>>>(aggws, wimg + WOUTT_OFF, b_out, out);
}
